// MoREGPT_2611340116570
// MI455X (gfx1250) — compile-verified, matrix operand order checked by reading, not run on hardware
//
#include <hip/hip_runtime.h>
#include <hip/hip_bf16.h>
#include <math.h>

typedef __attribute__((ext_vector_type(16))) _Float16 v16h;
typedef __attribute__((ext_vector_type(8)))  _Float16 v8h;
typedef __attribute__((ext_vector_type(8)))  float    v8f;

#define VOCAB 50257
#define DMODEL 768
#define NEXP 4
#define RRANK 32
#define BB 4
#define TT 512
#define MTOK (BB*TT)   // 2048
#define KCHUNK 128     // K staging chunk for A tile in LDS

// ---------------- K0: fp32 -> f16 convert of embedding table ----------------
__global__ __launch_bounds__(256) void cvt_f16_kernel(const float* __restrict__ in,
                                                      _Float16* __restrict__ out, int n) {
    int base = (blockIdx.x * 256 + threadIdx.x) * 4;
    if (base + 3 < n) {
        float4 v = *(const float4*)&in[base];
        out[base + 0] = (_Float16)v.x;
        out[base + 1] = (_Float16)v.y;
        out[base + 2] = (_Float16)v.z;
        out[base + 3] = (_Float16)v.w;
    } else {
        for (int i = base; i < n; ++i) out[i] = (_Float16)in[i];
    }
}

// ------- K1: per-token embedding gather + routing argmax + proj = x@W_in -----
__global__ __launch_bounds__(256) void gather_route_kernel(
    const int* __restrict__ idx, const float* __restrict__ W_emb,
    const float* __restrict__ G, const float* __restrict__ W_in,
    float* __restrict__ x, float* __restrict__ proj, int* __restrict__ winner) {
    const int tok = blockIdx.x;          // tok = b*T + t
    const int tid = threadIdx.x;
    __shared__ float xs[DMODEL];
    __shared__ float sc[NEXP][256];

    const int id = idx[tok];
    const float* src = W_emb + (size_t)id * DMODEL;
    float s0 = 0.f, s1 = 0.f, s2 = 0.f, s3 = 0.f;
    for (int d = tid; d < DMODEL; d += 256) {
        float v = src[d];
        xs[d] = v;
        x[(size_t)tok * DMODEL + d] = v;
        s0 += v * G[0 * DMODEL + d];
        s1 += v * G[1 * DMODEL + d];
        s2 += v * G[2 * DMODEL + d];
        s3 += v * G[3 * DMODEL + d];
    }
    sc[0][tid] = s0; sc[1][tid] = s1; sc[2][tid] = s2; sc[3][tid] = s3;
    __syncthreads();
    for (int off = 128; off > 0; off >>= 1) {
        if (tid < off) {
            sc[0][tid] += sc[0][tid + off];
            sc[1][tid] += sc[1][tid + off];
            sc[2][tid] += sc[2][tid + off];
            sc[3][tid] += sc[3][tid + off];
        }
        __syncthreads();
    }
    if (tid == 0) {
        float best = sc[0][0]; int be = 0;
        for (int e = 1; e < NEXP; ++e) if (sc[e][0] > best) { best = sc[e][0]; be = e; }
        winner[tok] = be;
    }
    // proj[tok][e][r] = sum_d xs[d] * W_in[e][d][r]
    if (tid < NEXP * RRANK) {
        const int e = tid >> 5, r = tid & 31;
        const float* wp = W_in + ((size_t)e * DMODEL) * RRANK + r;
        float acc = 0.f;
        for (int d = 0; d < DMODEL; ++d) acc += xs[d] * wp[(size_t)d * RRANK];
        proj[((size_t)tok * NEXP + e) * RRANK + r] = acc;
    }
}

// ----- K2: sequential recurrence per batch (winner-only state updates) -------
__global__ __launch_bounds__(32) void scan_rec_kernel(
    const float* __restrict__ proj, const float* __restrict__ W_rec,
    const int* __restrict__ winner, float* __restrict__ cand_sel) {
    const int b = blockIdx.x;
    const int s = threadIdx.x;           // 0..31
    __shared__ float wrec[NEXP * RRANK * RRANK];  // 16KB
    __shared__ float st[NEXP * RRANK];
    for (int i = s; i < NEXP * RRANK * RRANK; i += 32) wrec[i] = W_rec[i];
    for (int e = 0; e < NEXP; ++e) st[e * RRANK + s] = 0.f;
    __syncthreads();
    for (int t = 0; t < TT; ++t) {
        const int tok = b * TT + t;
        const int e = winner[tok];
        float acc = proj[((size_t)tok * NEXP + e) * RRANK + s];
        const float* wr = &wrec[e * RRANK * RRANK];
        const float* se = &st[e * RRANK];
#pragma unroll
        for (int r = 0; r < RRANK; ++r) acc += se[r] * wr[r * RRANK + s];
        float c = tanhf(acc);
        cand_sel[(size_t)tok * RRANK + s] = c;
        __syncthreads();
        st[e * RRANK + s] = c;
        __syncthreads();
    }
}

// ----- K3: y = cand@W_out[winner] + x ; LayerNorm ; emit h as f16 ------------
__global__ __launch_bounds__(256) void out_ln_kernel(
    const float* __restrict__ x, const float* __restrict__ cand_sel,
    const int* __restrict__ winner, const float* __restrict__ W_out,
    const float* __restrict__ gamma, const float* __restrict__ beta,
    _Float16* __restrict__ h16) {
    const int tok = blockIdx.x;
    const int tid = threadIdx.x;
    __shared__ float cs[RRANK];
    __shared__ float red[256];
    if (tid < RRANK) cs[tid] = cand_sel[(size_t)tok * RRANK + tid];
    __syncthreads();
    const int e = winner[tok];
    float y[3];
    float sum = 0.f, sq = 0.f;
#pragma unroll
    for (int j = 0; j < 3; ++j) {
        const int d = tid + j * 256;
        float acc = x[(size_t)tok * DMODEL + d];
        const float* wo = W_out + ((size_t)e * RRANK) * DMODEL + d;
#pragma unroll
        for (int r = 0; r < RRANK; ++r) acc += cs[r] * wo[(size_t)r * DMODEL];
        y[j] = acc; sum += acc; sq += acc * acc;
    }
    red[tid] = sum; __syncthreads();
    for (int off = 128; off > 0; off >>= 1) { if (tid < off) red[tid] += red[tid + off]; __syncthreads(); }
    const float mean = red[0] * (1.0f / DMODEL);
    __syncthreads();
    red[tid] = sq; __syncthreads();
    for (int off = 128; off > 0; off >>= 1) { if (tid < off) red[tid] += red[tid + off]; __syncthreads(); }
    const float var = red[0] * (1.0f / DMODEL) - mean * mean;
    const float inv = rsqrtf(var + 1e-5f);
#pragma unroll
    for (int j = 0; j < 3; ++j) {
        const int d = tid + j * 256;
        h16[(size_t)tok * DMODEL + d] = (_Float16)((y[j] - mean) * inv * gamma[d] + beta[d]);
    }
}

// ----- K4: logits = h @ W_emb^T via v_wmma_f32_16x16x32_f16 ------------------
// Block: 256 threads (8 waves). Block tile = 128(M) x 128(N).
// Wave (mw 0..3, nw 0..1) computes a 32(M) x 64(N) tile: 8 accumulators.
// A staged in LDS in K-chunks of 128 (32KB), shared by all waves.
// Per K-step: 2 A frags (LDS) + 4 B frags (global) feed 8 WMMAs.
__global__ __launch_bounds__(256) void head_gemm_kernel(
    const _Float16* __restrict__ h16, const _Float16* __restrict__ emb16,
    float* __restrict__ out) {
    __shared__ _Float16 As[128 * KCHUNK];   // 32KB
    const int mbase = blockIdx.y * 128;
    const int nbase = blockIdx.x * 128;
    const int tid  = threadIdx.x;
    const int lane = tid & 31;
    const int wave = tid >> 5;
    const int mw = wave & 3;                // 4 M-groups of 32 rows
    const int nw = wave >> 2;               // 2 N-groups of 64 cols
    const int frow  = lane & 15;            // M row (A) / N row (B) within subtile
    const int khalf = (lane >> 4) ? 8 : 0;  // per-lane K sub-offset

    // clamped B row pointers for the 4 N-subtiles (guarded stores handle tail)
    const _Float16* bp[4];
#pragma unroll
    for (int ns = 0; ns < 4; ++ns) {
        int n = nbase + nw * 64 + ns * 16 + frow;
        if (n >= VOCAB) n = VOCAB - 1;
        bp[ns] = emb16 + (size_t)n * DMODEL;
    }

    const v8f vzero = {};
    v8f acc[2][4];
#pragma unroll
    for (int ms = 0; ms < 2; ++ms)
#pragma unroll
        for (int ns = 0; ns < 4; ++ns) acc[ms][ns] = vzero;

    for (int k0 = 0; k0 < DMODEL; k0 += KCHUNK) {
        __syncthreads();
        // cooperative A-chunk load: 128 rows x KCHUNK halves (8-half chunks)
#pragma unroll
        for (int i = 0; i < (128 * KCHUNK / 8) / 256; ++i) {
            const int h8 = (i * 256 + tid) * 8;
            const int row = h8 / KCHUNK, col = h8 % KCHUNK;
            *(v8h*)&As[h8] = *(const v8h*)&h16[(size_t)(mbase + row) * DMODEL + k0 + col];
        }
        __syncthreads();
#pragma unroll
        for (int kk = 0; kk < KCHUNK; kk += 32) {
            v16h a[2], b[4];
#pragma unroll
            for (int ms = 0; ms < 2; ++ms) {
                const _Float16* ap = &As[(mw * 32 + ms * 16 + frow) * KCHUNK + kk + khalf];
                v8h lo = *(const v8h*)ap;
                v8h hi = *(const v8h*)(ap + 16);
                a[ms] = __builtin_shufflevector(lo, hi, 0,1,2,3,4,5,6,7,8,9,10,11,12,13,14,15);
            }
#pragma unroll
            for (int ns = 0; ns < 4; ++ns) {
                const _Float16* q = bp[ns] + k0 + kk + khalf;
                v8h lo = *(const v8h*)q;
                v8h hi = *(const v8h*)(q + 16);
                b[ns] = __builtin_shufflevector(lo, hi, 0,1,2,3,4,5,6,7,8,9,10,11,12,13,14,15);
            }
#pragma unroll
            for (int ms = 0; ms < 2; ++ms)
#pragma unroll
                for (int ns = 0; ns < 4; ++ns)
                    acc[ms][ns] = __builtin_amdgcn_wmma_f32_16x16x32_f16(
                        false, a[ms], false, b[ns], (short)0, acc[ms][ns], false, false);
        }
    }

    // C layout: VGPR j -> m = j + 8*(lane>=16); n = lane&15
    const int mlo = (lane >> 4) ? 8 : 0;
#pragma unroll
    for (int ms = 0; ms < 2; ++ms) {
        const int mrow = mbase + mw * 32 + ms * 16 + mlo;
#pragma unroll
        for (int ns = 0; ns < 4; ++ns) {
            const int ncol = nbase + nw * 64 + ns * 16 + (lane & 15);
            if (ncol < VOCAB) {
#pragma unroll
                for (int j = 0; j < 8; ++j)
                    out[(size_t)(mrow + j) * VOCAB + ncol] = acc[ms][ns][j];
            }
        }
    }
}

extern "C" void kernel_launch(void* const* d_in, const int* in_sizes, int n_in,
                              void* d_out, int out_size, void* d_ws, size_t ws_size,
                              hipStream_t stream) {
    const int*   idx    = (const int*)d_in[0];
    const float* W_emb  = (const float*)d_in[1];
    const float* G      = (const float*)d_in[2];
    const float* W_in   = (const float*)d_in[3];
    const float* W_rec  = (const float*)d_in[4];
    const float* W_out  = (const float*)d_in[5];
    const float* gamma  = (const float*)d_in[6];
    const float* beta   = (const float*)d_in[7];
    float* out = (float*)d_out;

    // workspace carve-up (256B aligned)
    auto align256 = [](size_t v) { return (v + 255) & ~(size_t)255; };
    char* ws = (char*)d_ws;
    size_t off = 0;
    _Float16* emb16 = (_Float16*)(ws + off); off = align256(off + (size_t)VOCAB * DMODEL * sizeof(_Float16));
    _Float16* h16   = (_Float16*)(ws + off); off = align256(off + (size_t)MTOK * DMODEL * sizeof(_Float16));
    float* x        = (float*)(ws + off);    off = align256(off + (size_t)MTOK * DMODEL * sizeof(float));
    float* proj     = (float*)(ws + off);    off = align256(off + (size_t)MTOK * NEXP * RRANK * sizeof(float));
    float* cand     = (float*)(ws + off);    off = align256(off + (size_t)MTOK * RRANK * sizeof(float));
    int* winner     = (int*)(ws + off);      off = align256(off + (size_t)MTOK * sizeof(int));
    (void)ws_size; (void)in_sizes; (void)n_in; (void)out_size;

    // K0: embedding table fp32 -> f16
    {
        int n = VOCAB * DMODEL;
        int blocks = (n / 4 + 255) / 256;
        cvt_f16_kernel<<<blocks, 256, 0, stream>>>(W_emb, emb16, n);
    }
    // K1: gather + routing + input projections
    gather_route_kernel<<<MTOK, 256, 0, stream>>>(idx, W_emb, G, W_in, x, proj, winner);
    // K2: sequential recurrence, one wave per batch element
    scan_rec_kernel<<<BB, 32, 0, stream>>>(proj, W_rec, winner, cand);
    // K3: expert output + residual + layernorm -> h (f16)
    out_ln_kernel<<<MTOK, 256, 0, stream>>>(x, cand, winner, W_out, gamma, beta, h16);
    // K4: WMMA tied head GEMM [2048 x 768] x [768 x 50257], 128x128 block tiles
    {
        dim3 grid((VOCAB + 127) / 128, MTOK / 128);
        head_gemm_kernel<<<grid, 256, 0, stream>>>(h16, emb16, out);
    }
}
